// MultivariablePrototypeLearningTrainLoopReal_70420283785369
// MI455X (gfx1250) — compile-verified, matrix operand order checked by reading, not run on hardware
//
#include <hip/hip_runtime.h>
#include <hip/hip_bf16.h>
#include <cstdint>

typedef __attribute__((ext_vector_type(16))) _Float16 v16h;
typedef __attribute__((ext_vector_type(8)))  float    v8f;
typedef __attribute__((ext_vector_type(4)))  unsigned uint4v;
typedef __attribute__((ext_vector_type(8)))  unsigned uint8v;

#define BB 512
#define TT 256
#define VV 8
#define HH 128
#define PP 64
#define CC 4
#define G4 512          // 4*H gate columns
#define KT 4            // k-tiles of 32 over H=128
#define BT 16           // batch rows per block
#define WIMG_HALFS 65536  // one swizzled 512x128 f16 image = 128KB

// ---- LDS partition (bytes) for the scan kernel ----
#define WHH_OFF   0
#define WHH_SZ    (KT*32*1024)              // 32 n-tiles x 4 k-tiles x 1KB = 128KB
#define AX_OFF    (WHH_OFF + WHH_SZ)
#define AX_SZ     (KT*1024)                 // 4KB
#define AH_OFF    (AX_OFF + AX_SZ)
#define AH_SZ     (KT*1024)                 // 4KB
#define GB_OFF    (AH_OFF + AH_SZ)
#define GB_SZ     (BT*G4*4)                 // 32KB gate preactivations
#define BIAS_OFF  (GB_OFF + GB_SZ)
#define BIAS_SZ   (G4*4)                    // 2KB
#define W0_OFF    (BIAS_OFF + BIAS_SZ)
#define W0_SZ     (G4*4)                    // 2KB
#define WIH_OFF   (W0_OFF + W0_SZ)
#define WIH_SZ    (KT*32*1024)              // 128KB
#define SM_L0     WIH_OFF                   // 176128 B (layer 0: no Wih frags)
#define SM_L12    (WIH_OFF + WIH_SZ)        // 307200 B <= 320KB WGP LDS

#define PROTO_SM  (32768 + 256*129*4)
#define FC1_SM    (16384 + 32768)

// ---- WMMA f16 16x16x32 per-lane fragment index mapping ----
__device__ __forceinline__ int kmap(int j, int pos, int hi) {
  int base = (j < 4) ? (hi ? 8 : 0) : (hi ? 24 : 16);
  return base + ((j & 3) << 1) + pos;
}

union Frag { v16h h; int4 q[2]; };
union H2   { _Float16 h[2]; unsigned u; };
union F4H8 { float4 f4; _Float16 h[8]; int4 q; };

__device__ __forceinline__ float sigm(float x)  { return 1.f / (1.f + __expf(-x)); }
__device__ __forceinline__ float tanh_(float x) { float e = __expf(2.f * x); return (e - 1.f) / (e + 1.f); }

// chunk cb (0..15) of 8 consecutive halves of row r -> contiguous 16B in frag image
__device__ __forceinline__ unsigned chunk_off(int cb, int r) {
  int kt = cb >> 2, sub = cb & 3;
  return (unsigned)(kt * 1024 + ((sub & 1) * 16 + r) * 32 + (sub >> 1) * 16);
}

// TDM: DMA a linear image of n8 8-byte elements from global into LDS at lds_off.
__device__ __forceinline__ void tdm_load_image(unsigned lds_off, const void* gsrc, unsigned n8) {
  unsigned long long ga = (unsigned long long)(uintptr_t)gsrc;
  uint4v g0;
  g0[0] = 1u;                                            // count=1 (valid), user mode
  g0[1] = lds_off;                                       // lds_addr
  g0[2] = (unsigned)(ga & 0xFFFFFFFFu);                  // global_addr[31:0]
  g0[3] = (unsigned)((ga >> 32) & 0x01FFFFFFu) | (2u << 30);  // addr[56:32] | type=2
  uint8v g1;
  g1[0] = 3u << 16;                                      // data_size=3 (8B), wg_mask=0
  g1[1] = (n8 & 0xFFFFu) << 16;                          // tensor_dim0[15:0]
  g1[2] = (n8 >> 16) | (1u << 16);                       // tensor_dim0[31:16], tensor_dim1=1
  g1[3] = (n8 & 0xFFFFu) << 16;                          // tile_dim0 (n8 <= 65535)
  g1[4] = 1u;                                            // tile_dim1=1
  g1[5] = n8;                                            // tensor_dim0_stride
  g1[6] = 0u;
  g1[7] = 0u;
  asm volatile("tensor_load_to_lds %0, %1" :: "s"(g0), "s"(g1) : "memory");
}

// Prep: convert each [512,128] fp32 weight into a pre-swizzled f16 fragment image
// (frag(nt,kt) at (nt*KT+kt)*1024B, lane-contiguous 32B) stored in workspace.
__global__ __launch_bounds__(256)
void swizzle_weights_kernel(const float* __restrict__ Whh0, const float* __restrict__ Wih1,
                            const float* __restrict__ Whh1, const float* __restrict__ Wih2,
                            const float* __restrict__ Whh2, _Float16* __restrict__ wimg) {
  const int im = blockIdx.x;               // 0..39 = mat*8 + v
  const int mat = im >> 3, v = im & 7;
  const float* W =
      (mat == 0 ? Whh0 : mat == 1 ? Wih1 : mat == 2 ? Whh1 : mat == 3 ? Wih2 : Whh2)
      + (size_t)v * G4 * HH;
  unsigned* dst = (unsigned*)(wimg + (size_t)im * WIMG_HALFS);
  for (int o = threadIdx.x; o < (G4 * HH) / 2; o += 256) {
    int f = o >> 8, r = o & 255, L = r >> 3, j = r & 7;
    int nt = f >> 2, kt = f & 3;
    int n = nt * 16 + (L & 15);
    int hi = (L >= 16) ? 1 : 0;
    H2 p;
    p.h[0] = (_Float16)W[n * HH + kt * 32 + kmap(j, 0, hi)];
    p.h[1] = (_Float16)W[n * HH + kt * 32 + kmap(j, 1, hi)];
    dst[f * 256 + L * 8 + j] = p.u;
  }
}

// One LSTM layer, full T-scan. Block = (variable, 16-row batch tile); 8 waves.
__global__ __launch_bounds__(256, 1)
void lstm_scan_kernel(int layer_kind,
                      const float*    __restrict__ data,      // [B,T,V] (layer 0)
                      const _Float16* __restrict__ ys_in,     // [V,T,B,H] f16 (layers 1/2)
                      _Float16*       __restrict__ ys_out,    // [V,T,B,H] f16 or null
                      const _Float16* __restrict__ wih_img,   // swizzled images [V] or null
                      const _Float16* __restrict__ whh_img,   // swizzled images [V]
                      const float*    __restrict__ biasAll,   // [V,512]
                      const float*    __restrict__ Wih0All,   // [V,512] or null
                      float*          __restrict__ enc_out)   // [V,B,H] or null
{
  extern __shared__ char smem[];
  const int tid = threadIdx.x;
  const int v  = blockIdx.x >> 5;
  const int b0 = (blockIdx.x & 31) * BT;
  const int wave = tid >> 5, lane = tid & 31;
  const int r = tid >> 4, cb = tid & 15;       // gate-phase ownership: row r, chunk cb
  const int hc0 = cb << 3;

  // ---- TDM: pull pre-swizzled weight fragment images straight into LDS ----
  if (wave == 0) {
    tdm_load_image(WHH_OFF + (unsigned)(uintptr_t)smem,
                   whh_img + (size_t)v * WIMG_HALFS, WHH_SZ / 8);
    if (layer_kind == 1)
      tdm_load_image(WIH_OFF + (unsigned)(uintptr_t)smem,
                     wih_img + (size_t)v * WIMG_HALFS, WIH_SZ / 8);
    __builtin_amdgcn_s_wait_tensorcnt(0);
  }
  for (int o = tid; o < G4; o += 256) {
    *(float*)(smem + BIAS_OFF + o * 4) = biasAll[(size_t)v * G4 + o];
    if (layer_kind == 0) *(float*)(smem + W0_OFF + o * 4) = Wih0All[(size_t)v * G4 + o];
  }
  for (int o = tid; o < AH_SZ / 16; o += 256) {
    int4 z = {0, 0, 0, 0};
    *(int4*)(smem + AH_OFF + o * 16) = z;
  }

  float cst[8];
  #pragma unroll
  for (int e = 0; e < 8; ++e) cst[e] = 0.f;

  __syncthreads();

  for (int t = 0; t < TT; ++t) {
    // ---- Phase 1: async-stage x-input A fragments (one b128 per thread) ----
    if (layer_kind == 1) {
      unsigned lds_off = ((unsigned)(uintptr_t)smem) + AX_OFF + chunk_off(cb, r);
      unsigned long long gsrc =
          (unsigned long long)(uintptr_t)(ys_in + (((size_t)v * TT + t) * BB + (b0 + r)) * HH + hc0);
      asm volatile("global_load_async_to_lds_b128 %0, %1, off"
                   :: "v"(lds_off), "v"(gsrc) : "memory");
      if (t + 1 < TT)
        __builtin_prefetch(ys_in + (((size_t)v * TT + t + 1) * BB + (b0 + r)) * HH + hc0, 0, 0);
      asm volatile("s_wait_asynccnt 0" ::: "memory");
    }
    __syncthreads();

    // ---- Phase 2: WMMA GEMMs -> gate preactivations (16 x 512) ----
    v8f acc[4];
    #pragma unroll
    for (int n = 0; n < 4; ++n)
      #pragma unroll
      for (int i = 0; i < 8; ++i) acc[n][i] = 0.f;

    #pragma unroll
    for (int ntl = 0; ntl < 4; ++ntl) {
      const int nt = wave * 4 + ntl;
      #pragma unroll
      for (int kt = 0; kt < KT; ++kt) {          // h @ Whh.T
        Frag a, b;
        const char* ap = smem + AH_OFF + kt * 1024 + lane * 32;
        a.q[0] = *(const int4*)ap; a.q[1] = *(const int4*)(ap + 16);
        const char* bp = smem + WHH_OFF + (nt * KT + kt) * 1024 + lane * 32;
        b.q[0] = *(const int4*)bp; b.q[1] = *(const int4*)(bp + 16);
        acc[ntl] = __builtin_amdgcn_wmma_f32_16x16x32_f16(
            false, a.h, false, b.h, (short)0, acc[ntl], false, false);
      }
      if (layer_kind == 1) {
        #pragma unroll
        for (int kt = 0; kt < KT; ++kt) {        // x @ Wih.T
          Frag a, b;
          const char* ap = smem + AX_OFF + kt * 1024 + lane * 32;
          a.q[0] = *(const int4*)ap; a.q[1] = *(const int4*)(ap + 16);
          const char* bp = smem + WIH_OFF + (nt * KT + kt) * 1024 + lane * 32;
          b.q[0] = *(const int4*)bp; b.q[1] = *(const int4*)(bp + 16);
          acc[ntl] = __builtin_amdgcn_wmma_f32_16x16x32_f16(
              false, a.h, false, b.h, (short)0, acc[ntl], false, false);
        }
      }
      #pragma unroll
      for (int vg = 0; vg < 8; ++vg) {
        int row = vg + ((lane >= 16) ? 8 : 0);
        int col = nt * 16 + (lane & 15);
        *(float*)(smem + GB_OFF + (row * G4 + col) * 4) = acc[ntl][vg];
      }
    }
    __syncthreads();

    // ---- Phase 3: gate nonlinearities + cell/hidden update ----
    float xv = 0.f;
    if (layer_kind == 0) xv = data[((size_t)(b0 + r) * TT + t) * VV + v];
    F4H8 hu;
    #pragma unroll
    for (int e = 0; e < 8; ++e) {
      int hc = hc0 + e;
      float gi = *(float*)(smem + GB_OFF + (r * G4 + hc) * 4)        + *(float*)(smem + BIAS_OFF + hc * 4);
      float gf = *(float*)(smem + GB_OFF + (r * G4 + 128 + hc) * 4)  + *(float*)(smem + BIAS_OFF + (128 + hc) * 4);
      float gg = *(float*)(smem + GB_OFF + (r * G4 + 256 + hc) * 4)  + *(float*)(smem + BIAS_OFF + (256 + hc) * 4);
      float go = *(float*)(smem + GB_OFF + (r * G4 + 384 + hc) * 4)  + *(float*)(smem + BIAS_OFF + (384 + hc) * 4);
      if (layer_kind == 0) {
        gi += xv * *(float*)(smem + W0_OFF + hc * 4);
        gf += xv * *(float*)(smem + W0_OFF + (128 + hc) * 4);
        gg += xv * *(float*)(smem + W0_OFF + (256 + hc) * 4);
        go += xv * *(float*)(smem + W0_OFF + (384 + hc) * 4);
      }
      float ig = sigm(gi), fg = sigm(gf), gc = tanh_(gg), og = sigm(go);
      cst[e] = fg * cst[e] + ig * gc;
      float hval = og * tanh_(cst[e]);
      hu.h[e] = (_Float16)hval;
      if (enc_out && t == TT - 1) enc_out[((size_t)v * BB + (b0 + r)) * HH + hc] = hval;
    }
    // one contiguous 16B chunk into the h fragment buffer + one 16B global store
    *(int4*)(smem + AH_OFF + chunk_off(cb, r)) = hu.q;
    if (ys_out)
      *(float4*)(ys_out + (((size_t)v * TT + t) * BB + (b0 + r)) * HH + hc0) = hu.f4;
    // next iteration's staging barrier separates these writes from the next GEMM
  }
}

// Prototype distances + softmax(1/dist) -> concat [B, V*P].
__global__ __launch_bounds__(256)
void proto_softmax_kernel(const float* __restrict__ enc,     // [V,B,H]
                          const float* __restrict__ protos,  // [V,P,H]
                          float* __restrict__ concat)        // [B, V*P]
{
  extern __shared__ char smem[];
  float* proS = (float*)smem;                    // [P*H]
  float* encS = (float*)(smem + 32768);          // [256*129] padded
  const int tid = threadIdx.x;
  const int v = blockIdx.x >> 1;
  const int b = (blockIdx.x & 1) * 256 + tid;
  for (int o = tid; o < PP * HH; o += 256) proS[o] = protos[(size_t)v * PP * HH + o];
  for (int k = 0; k < HH; ++k) encS[tid * 129 + k] = enc[((size_t)v * BB + b) * HH + k];
  __syncthreads();
  float m = -1e30f, s = 0.f;
  for (int p = 0; p < PP; ++p) {
    float d = 0.f;
    for (int k = 0; k < HH; ++k) {
      float df = encS[tid * 129 + k] - proS[p * HH + k];
      d += df * df;
    }
    float l = rsqrtf(d);
    float nm = fmaxf(m, l);
    s = s * __expf(m - nm) + __expf(l - nm);
    m = nm;
  }
  float rs = 1.f / s;
  for (int p = 0; p < PP; ++p) {
    float d = 0.f;
    for (int k = 0; k < HH; ++k) {
      float df = encS[tid * 129 + k] - proS[p * HH + k];
      d += df * df;
    }
    float l = rsqrtf(d);
    concat[(size_t)b * (VV * PP) + v * PP + p] = __expf(l - m) * rs;
  }
}

// h1 = relu(concat @ W1.T + b1), WMMA f16. Block: 16 rows x 512 cols; K=512.
__global__ __launch_bounds__(256)
void fc1_kernel(const float* __restrict__ concat,  // [B,512]
                const float* __restrict__ W1,      // [1024,512]
                const float* __restrict__ b1,      // [1024]
                _Float16* __restrict__ h1)         // [B,1024] f16
{
  extern __shared__ char smem[];
  const int tid = threadIdx.x;
  const int m0 = (blockIdx.x >> 1) * 16;
  const int nb = blockIdx.x & 1;
  const int wave = tid >> 5, lane = tid & 31;

  for (int o = tid; o < 16 * 256; o += 256) {
    int kt = o >> 8, rr = o & 255, L = rr >> 3, j = rr & 7;
    int row = L & 15, hi = (L >= 16) ? 1 : 0;
    H2 p;
    p.h[0] = (_Float16)concat[(size_t)(m0 + row) * 512 + kt * 32 + kmap(j, 0, hi)];
    p.h[1] = (_Float16)concat[(size_t)(m0 + row) * 512 + kt * 32 + kmap(j, 1, hi)];
    *(unsigned*)(smem + kt * 1024 + L * 32 + j * 4) = p.u;
  }

  v8f acc[4];
  #pragma unroll
  for (int n = 0; n < 4; ++n)
    #pragma unroll
    for (int i = 0; i < 8; ++i) acc[n][i] = 0.f;

  for (int kt = 0; kt < 16; ++kt) {
    __syncthreads();
    for (int o = tid; o < 32 * 256; o += 256) {
      int ntf = o >> 8, rr = o & 255, L = rr >> 3, j = rr & 7;
      int n = nb * 512 + ntf * 16 + (L & 15);
      int hi = (L >= 16) ? 1 : 0;
      H2 p;
      p.h[0] = (_Float16)W1[(size_t)n * 512 + kt * 32 + kmap(j, 0, hi)];
      p.h[1] = (_Float16)W1[(size_t)n * 512 + kt * 32 + kmap(j, 1, hi)];
      *(unsigned*)(smem + 16384 + ntf * 1024 + L * 32 + j * 4) = p.u;
    }
    __syncthreads();
    Frag a;
    const char* ap = smem + kt * 1024 + lane * 32;
    a.q[0] = *(const int4*)ap; a.q[1] = *(const int4*)(ap + 16);
    #pragma unroll
    for (int ntl = 0; ntl < 4; ++ntl) {
      Frag b;
      const char* bp = smem + 16384 + (wave * 4 + ntl) * 1024 + lane * 32;
      b.q[0] = *(const int4*)bp; b.q[1] = *(const int4*)(bp + 16);
      acc[ntl] = __builtin_amdgcn_wmma_f32_16x16x32_f16(
          false, a.h, false, b.h, (short)0, acc[ntl], false, false);
    }
  }
  #pragma unroll
  for (int ntl = 0; ntl < 4; ++ntl) {
    int n = nb * 512 + (wave * 4 + ntl) * 16 + (lane & 15);
    float bv = b1[n];
    #pragma unroll
    for (int vg = 0; vg < 8; ++vg) {
      int row = vg + ((lane >= 16) ? 8 : 0);
      float hv = fmaxf(acc[ntl][vg] + bv, 0.f);
      h1[(size_t)(m0 + row) * 1024 + n] = (_Float16)hv;
    }
  }
}

// agg = h1 @ W2.T + b2 ; [512,1024] x [1024,4] -> tiny dot kernel.
__global__ __launch_bounds__(256)
void fc2_kernel(const _Float16* __restrict__ h1, const float* __restrict__ W2,
                const float* __restrict__ b2, float* __restrict__ agg)
{
  int g = blockIdx.x * 256 + threadIdx.x;
  if (g >= BB * CC) return;
  int b = g >> 2, c = g & 3;
  float acc = b2[c];
  for (int k = 0; k < 1024; ++k)
    acc += (float)h1[(size_t)b * 1024 + k] * W2[(size_t)c * 1024 + k];
  agg[g] = acc;
}

extern "C" void kernel_launch(void* const* d_in, const int* in_sizes, int n_in,
                              void* d_out, int out_size, void* d_ws, size_t ws_size,
                              hipStream_t stream) {
  (void)in_sizes; (void)n_in; (void)out_size; (void)ws_size;
  const float* data   = (const float*)d_in[0];
  const float* Wih0   = (const float*)d_in[1];
  const float* Whh0   = (const float*)d_in[2];
  const float* bias0  = (const float*)d_in[3];
  const float* Wih1   = (const float*)d_in[4];
  const float* Whh1   = (const float*)d_in[5];
  const float* bias1  = (const float*)d_in[6];
  const float* Wih2   = (const float*)d_in[7];
  const float* Whh2   = (const float*)d_in[8];
  const float* bias2  = (const float*)d_in[9];
  const float* protos = (const float*)d_in[10];
  const float* W1     = (const float*)d_in[11];
  const float* b1     = (const float*)d_in[12];
  const float* W2     = (const float*)d_in[13];
  const float* b2     = (const float*)d_in[14];

  char* ws = (char*)d_ws;
  const size_t ys_bytes = (size_t)VV * TT * BB * HH * 2;     // 256 MB f16 each
  _Float16* ys_a = (_Float16*)ws;
  _Float16* ys_b = (_Float16*)(ws + ys_bytes);
  float*    enc  = (float*)(ws + 2 * ys_bytes);              // [V,B,H] f32
  _Float16* h1   = (_Float16*)(ws + 2 * ys_bytes + (size_t)VV * BB * HH * 4);
  _Float16* wimg = (_Float16*)(ws + 2 * ys_bytes + (size_t)VV * BB * HH * 4
                               + (size_t)BB * 1024 * 2);     // 40 x 128KB images

  _Float16* img_whh0 = wimg + (size_t)0  * VV * WIMG_HALFS;
  _Float16* img_wih1 = wimg + (size_t)1  * VV * WIMG_HALFS;
  _Float16* img_whh1 = wimg + (size_t)2  * VV * WIMG_HALFS;
  _Float16* img_wih2 = wimg + (size_t)3  * VV * WIMG_HALFS;
  _Float16* img_whh2 = wimg + (size_t)4  * VV * WIMG_HALFS;

  float* agg    = (float*)d_out;                             // [B,C]
  float* concat = agg + BB * CC;                             // [B,V*P]

  dim3 blk(256);
  swizzle_weights_kernel<<<dim3(40), blk, 0, stream>>>(Whh0, Wih1, Whh1, Wih2, Whh2, wimg);
  lstm_scan_kernel<<<dim3(VV * 32), blk, SM_L0, stream>>>(
      0, data, nullptr, ys_a, nullptr, img_whh0, bias0, Wih0, nullptr);
  lstm_scan_kernel<<<dim3(VV * 32), blk, SM_L12, stream>>>(
      1, nullptr, ys_a, ys_b, img_wih1, img_whh1, bias1, nullptr, nullptr);
  lstm_scan_kernel<<<dim3(VV * 32), blk, SM_L12, stream>>>(
      1, nullptr, ys_b, nullptr, img_wih2, img_whh2, bias2, nullptr, enc);
  proto_softmax_kernel<<<dim3(VV * 2), blk, PROTO_SM, stream>>>(enc, protos, concat);
  fc1_kernel<<<dim3(64), blk, FC1_SM, stream>>>(concat, W1, b1, h1);
  fc2_kernel<<<dim3(8), blk, 0, stream>>>(h1, W2, b2, agg);
}
